// DenseDilatedKnnGraph_13546326851640
// MI455X (gfx1250) — matrix operand order repaired and checked by reading, not compiled, not run on hardware
//
#include <hip/hip_runtime.h>
#include <hip/hip_bf16.h>

// ---------------------------------------------------------------------------
// DenseDilatedKnnGraph for MI455X (gfx1250, wave32, WMMA)
//   x: [B=2, C=192, N=8192, 1] f32
//   out: edge_index [2, B, N, 9] int32  (first half = nn idx, second = center)
// ---------------------------------------------------------------------------

#define B_  2
#define C_  192
#define N_  8192
#define K_  9
#define NWAVES 8
#define NCHUNK (C_ / 32)   // 6 K-chunks of 32 for v_wmma_f32_16x16x32_f16
#define TPAD 20            // tile row pad: 20 floats = 80B, keeps 16B alignment

typedef __attribute__((ext_vector_type(8)))  _Float16 v8h;
typedef __attribute__((ext_vector_type(16))) _Float16 v16h;
typedef __attribute__((ext_vector_type(8)))  float    v8f;
typedef __attribute__((ext_vector_type(4)))  float    v4f;
typedef unsigned long long u64;
typedef unsigned int       u32;

// -------------------------- normalize + pack f16 ---------------------------
// x[b][c][n] -> pts[b][n][c] = (f16)(x / max(||x||,eps)), row-major C.
__global__ __launch_bounds__(256) void knn_normalize_kernel(
    const float* __restrict__ x, _Float16* __restrict__ pts) {
  int t = blockIdx.x * 256 + threadIdx.x;      // 0 .. B_*N_-1
  int b = t >> 13;                             // / N_
  int n = t & (N_ - 1);
  const float* xb = x + (size_t)b * C_ * N_ + n;

  float s = 0.0f;
#pragma unroll 4
  for (int c = 0; c < C_; ++c) {
    float v = xb[(size_t)c * N_];
    s += v * v;
  }
  float scale = 1.0f / fmaxf(sqrtf(s), 1e-12f);

  unsigned int* p = (unsigned int*)(pts + (size_t)t * C_);
#pragma unroll 4
  for (int c = 0; c < C_; c += 2) {
    float v0 = xb[(size_t)c * N_] * scale;
    float v1 = xb[(size_t)(c + 1) * N_] * scale;
    union { _Float16 h[2]; unsigned int u; } pk;
    pk.h[0] = (_Float16)v0;
    pk.h[1] = (_Float16)v1;
    p[c >> 1] = pk.u;
  }
}

// ------------------------------ top-k helpers ------------------------------
// Sortable u64 key: [63:32] = order-preserving transform of f32 distance,
// [31:0] = column index.  Ascending u64 order == ascending (dist, idx),
// which reproduces jax.lax.top_k's smallest-index tie-breaking.
__device__ __forceinline__ u64 make_key(float d, int idx) {
  u32 bits = __float_as_uint(d);
  u32 tr   = bits ^ ((u32)((int)bits >> 31) | 0x80000000u);
  return ((u64)tr << 32) | (u32)idx;
}

// Guarded, fully-unrolled bubble insertion on u64 keys; compiles to a
// v_min_u64/v_max_u64 sorting-network pass (2 VALU per slot).
__device__ __forceinline__ void kinsert(u64 (&bk)[K_], u64 key) {
  if (key >= bk[K_ - 1]) return;
  u64 c = key;
#pragma unroll
  for (int j = 0; j < K_; ++j) {
    u64 t = bk[j];
    bool l = c < t;
    bk[j] = l ? c : t;
    c     = l ? t : c;
  }
}

// --------------------------- WMMA KNN main kernel --------------------------
// Block = 16 rows of one batch; 8 waves stripe the 512 column tiles.
__global__ __launch_bounds__(256) void knn_wmma_kernel(
    const _Float16* __restrict__ pts, int* __restrict__ out) {
  const int lane = threadIdx.x & 31;
  const int wave = threadIdx.x >> 5;
  const int m    = lane & 15;   // row/col within 16-tile
  const int h    = lane >> 4;   // lane half

  const int bt       = blockIdx.x;          // 0 .. B_*(N_/16)-1
  const int b        = bt >> 9;             // / (N_/16)
  const int rt       = bt & ((N_ / 16) - 1);
  const int row_base = rt * 16;

  const _Float16* base = pts + (size_t)b * N_ * C_;

  __shared__ float tile[NWAVES][16][TPAD];     // per-wave 16x16 transpose
  __shared__ u64   candK[16][NWAVES * 18];     // 144 candidates / row

  // ---- A fragments: 16 rows x 192, resident in VGPRs (6 x v16h = 48 VGPRs)
  // 16-bit A 16x32 layout: lane(m,h): elems 0..7 -> K = 8h+0..7,
  //                                   elems 8..15 -> K = 16+8h+0..7
  v16h afrag[NCHUNK];
  {
    const _Float16* arow = base + (size_t)(row_base + m) * C_;
#pragma unroll
    for (int kc = 0; kc < NCHUNK; ++kc) {
      const int k0 = kc * 32;
      v8h lo = *(const v8h*)(arow + k0 + h * 8);
      v8h hi = *(const v8h*)(arow + k0 + 16 + h * 8);
      afrag[kc] = __builtin_shufflevector(lo, hi, 0, 1, 2, 3, 4, 5, 6, 7,
                                          8, 9, 10, 11, 12, 13, 14, 15);
    }
  }

  u64 bk[K_];
#pragma unroll
  for (int j = 0; j < K_; ++j) bk[j] = ~0ull;

  // ---- stripe over column tiles
  for (int ct = wave; ct < N_ / 16; ct += NWAVES) {
    const int col_base = ct * 16;
    const _Float16* brow = base + (size_t)(col_base + m) * C_;

    // 16-bit B 32x16 layout: lane(n=m,h): elems 0..7 -> K = 8h+0..7, elems 8..15 -> K = 16+8h+0..7
    v16h bfrag[NCHUNK];
#pragma unroll
    for (int kc = 0; kc < NCHUNK; ++kc) {
      const int k0 = kc * 32;
      v8h l0 = *(const v8h*)(brow + k0 + h * 8);
      v8h l1 = *(const v8h*)(brow + k0 + 16 + h * 8);
      bfrag[kc] = __builtin_shufflevector(l0, l1, 0, 1, 2, 3, 4, 5, 6, 7,
                                          8, 9, 10, 11, 12, 13, 14, 15);
    }

    // Two independent accumulator chains (even/odd chunks) so the scheduler
    // can keep two WMMAs in flight and overlap loads with the other chain.
    v8f acc0 = {}, acc1 = {};
#pragma unroll
    for (int kc = 0; kc < NCHUNK; kc += 2) {
      acc0 = __builtin_amdgcn_wmma_f32_16x16x32_f16(
          false, afrag[kc],     false, bfrag[kc],     (short)0, acc0,
          false, false);
      acc1 = __builtin_amdgcn_wmma_f32_16x16x32_f16(
          false, afrag[kc + 1], false, bfrag[kc + 1], (short)0, acc1,
          false, false);
    }
    v8f acc = acc0 + acc1;

    // transpose accumulator through this wave's private LDS tile
    // acc[v] @ lane(m,h) = inner[row = v + 8h][col = m]
#pragma unroll
    for (int v = 0; v < 8; ++v) tile[wave][v + 8 * h][m] = acc[v];

    // scan: lane(m,h) owns row m, columns h*8 .. h*8+7 (two ds_load_b128)
    const v4f* rowp = (const v4f*)&tile[wave][m][h * 8];
    v4f c0 = rowp[0];
    v4f c1 = rowp[1];
    float vals[8] = {c0.x, c0.y, c0.z, c0.w, c1.x, c1.y, c1.z, c1.w};
#pragma unroll
    for (int i = 0; i < 8; ++i) {
      // dist = 2 - 2*inner is monotone in -inner (all points unit-norm)
      kinsert(bk, make_key(-vals[i], col_base + h * 8 + i));
    }
  }

  // ---- dump per-lane top-9 into the merge buffer
#pragma unroll
  for (int j = 0; j < K_; ++j)
    candK[m][wave * 18 + h * K_ + j] = bk[j];
  __syncthreads();

  // ---- final merge: 16 threads, one row each
  if (threadIdx.x < 16) {
    const int r = threadIdx.x;
    u64 fk[K_];
#pragma unroll
    for (int j = 0; j < K_; ++j) fk[j] = ~0ull;
    for (int j = 0; j < NWAVES * 18; ++j)
      kinsert(fk, candK[r][j]);

    const int n = row_base + r;
    const size_t o0 = ((size_t)b * N_ + n) * K_;
#pragma unroll
    for (int j = 0; j < K_; ++j) {
      out[o0 + j]                        = (int)(u32)fk[j];  // edge_index[0]
      out[(size_t)B_ * N_ * K_ + o0 + j] = n;                // edge_index[1]
    }
  }
}

// ------------------------------- launcher ----------------------------------
extern "C" void kernel_launch(void* const* d_in, const int* in_sizes, int n_in,
                              void* d_out, int out_size, void* d_ws,
                              size_t ws_size, hipStream_t stream) {
  const float* x   = (const float*)d_in[0];
  int* out         = (int*)d_out;
  _Float16* pts    = (_Float16*)d_ws;   // B_*N_*C_ halves = 6 MB scratch

  knn_normalize_kernel<<<(B_ * N_) / 256, 256, 0, stream>>>(x, pts);
  knn_wmma_kernel<<<B_ * (N_ / 16), 256, 0, stream>>>(pts, out);
}
